// GCN_24584392802582
// MI455X (gfx1250) — matrix operand order repaired and checked by reading, not compiled, not run on hardware
//
#include <hip/hip_runtime.h>
#include <stdint.h>

// GCN on MI455X (gfx1250): bf16 WMMA GEMM (LDS-staged B, explicit double-buffered
// A and B fragments) + atomic scatter aggregation.
// ws layout (~77.2 MB): dinv[N] f32 | Hbf[N*256] bf16 | Wbf[256*256] bf16 | G[N*256] f32
// d_out doubles as the per-layer aggregation buffer Y.

typedef __bf16 v16bf __attribute__((ext_vector_type(16)));
typedef __bf16 bf16x2 __attribute__((ext_vector_type(2)));
typedef float  v8f    __attribute__((ext_vector_type(8)));

#define DFEAT 256

// ---------- degree / normalization ----------
__global__ void deg_init_kernel(float* __restrict__ deg, int n) {
    int i = blockIdx.x * blockDim.x + threadIdx.x;
    if (i < n) deg[i] = 1.0f;                     // self-loop contributes 1
}

__global__ void deg_accum_kernel(const int* __restrict__ col, float* __restrict__ deg, int e) {
    int i = blockIdx.x * blockDim.x + threadIdx.x;
    if (i < e) atomicAdd(&deg[col[i]], 1.0f);
}

__global__ void deg_rsqrt_kernel(float* __restrict__ deg, int n) {
    int i = blockIdx.x * blockDim.x + threadIdx.x;
    if (i < n) deg[i] = rsqrtf(deg[i]);           // deg >= 1 always (self loop)
}

// ---------- f32 -> bf16 conversion (optionally fused ReLU) ----------
__global__ void cvt_bf16_kernel(const float* __restrict__ in, __bf16* __restrict__ out,
                                int n, int do_relu) {
    int i = blockIdx.x * blockDim.x + threadIdx.x;
    if (i < n) {
        float v = in[i];
        if (do_relu) v = fmaxf(v, 0.0f);
        out[i] = (__bf16)v;
    }
}

// A 16x32 bf16 fragment: VGPR v -> K = 2v (+16 upper lanes)
static __device__ __forceinline__ v16bf load_afrag(const __bf16* __restrict__ arow,
                                                   int kk, int half) {
    v16bf af;
#pragma unroll
    for (int v = 0; v < 8; ++v) {
        int kA = kk + 2 * v + half * 16;
        bf16x2 pa = *(const bf16x2*)(arow + kA);
        af[2 * v]     = pa[0];
        af[2 * v + 1] = pa[1];
    }
    return af;
}

// ---------- WMMA GEMM: C[n,j] = sum_k A[n,k] * W[j,k]  (i.e. A @ W^T) ----------
// Block = 8 waves = 8 M-tiles (128 rows) x one N-group (64 cols).
// The 64x256 bf16 W-slice (32 KB) is staged in LDS once per block. K-loop keeps
// two in-flight buffers of the A fragment (global) and of all 4 B fragments (LDS)
// so every WMMA consumes loads issued one full K-step earlier.
__global__ __launch_bounds__(256)
void gemm_wmma_bf16_kernel(const __bf16* __restrict__ A,   // Nrows x 256, row-major
                           const __bf16* __restrict__ W,   // 256 x 256, row-major
                           float* __restrict__ C,          // Nrows x 256
                           int Nrows) {
    __shared__ __bf16 Bs[64 * DFEAT];              // 32 KB

    const int tid  = threadIdx.x;
    const int ng   = blockIdx.x & 3;               // N-group (64 cols)
    const int bm   = blockIdx.x >> 2;              // M-block (8 tiles)
    const int wv   = tid >> 5;                     // wave in block -> M-tile
    const int lane = tid & 31;
    const int half = lane >> 4;
    const int l15  = lane & 15;

    // ---- cooperative staging of the W slice: rows [ng*64, ng*64+64) ----
    {
        const float4* wsrc = (const float4*)(W + (size_t)ng * 64 * DFEAT);
        float4* bdst = (float4*)Bs;
#pragma unroll
        for (int i = 0; i < 8; ++i)                // 2048 b128 chunks / 256 threads
            bdst[tid + i * 256] = wsrc[tid + i * 256];
    }
    __syncthreads();

    const int mt = bm * 8 + wv;                    // M-tile index
    if ((mt + 1) * 16 > Nrows) return;             // wave-uniform tail exit (after barrier)

    const __bf16* arow = A + (size_t)(mt * 16 + l15) * DFEAT;   // A-frag: M = lane&15
    // B fragment for tile t at K-step kk: 32 contiguous bytes of the LDS row
    const __bf16* bbase = Bs + (size_t)l15 * DFEAT + half * 16;

    v8f acc[4] = {{}, {}, {}, {}};
    v16bf af[2];
    v16bf bfr[2][4];

    // prologue: fill buffer 0 for kk = 0
    af[0] = load_afrag(arow, 0, half);
#pragma unroll
    for (int t = 0; t < 4; ++t)
        bfr[0][t] = *(const v16bf*)(bbase + (size_t)t * 16 * DFEAT);

#pragma unroll
    for (int kk = 0; kk < DFEAT; kk += 32) {
        const int cur = (kk >> 5) & 1;
        const int nxt = cur ^ 1;
        // issue next K-step's loads before consuming current buffers
        if (kk + 32 < DFEAT) {
            af[nxt] = load_afrag(arow, kk + 32, half);
#pragma unroll
            for (int t = 0; t < 4; ++t)
                bfr[nxt][t] = *(const v16bf*)(bbase + (size_t)t * 16 * DFEAT + kk + 32);
        }
#pragma unroll
        for (int t = 0; t < 4; ++t) {
            acc[t] = __builtin_amdgcn_wmma_f32_16x16x32_bf16(
                /*neg_a=*/false, af[cur], /*neg_b=*/false, bfr[cur][t],
                /*c_mod=*/(short)0, acc[t], /*reuse_a=*/false, /*reuse_b=*/false);
        }
    }

    // C/D 16x16 f32 layout: lane&15 = N, VGPR v -> M = v + 8*half
#pragma unroll
    for (int t = 0; t < 4; ++t) {
        float* crow = C + (size_t)(ng * 64 + t * 16 + l15);
#pragma unroll
        for (int v = 0; v < 8; ++v) {
            crow[(size_t)(mt * 16 + v + 8 * half) * DFEAT] = acc[t][v];
        }
    }
}

// ---------- Y = bias + dinv^2 * G  (self-loop term, also initializes Y) ----------
__global__ void init_out_kernel(const float* __restrict__ G, const float* __restrict__ dinv,
                                const float* __restrict__ bias, float* __restrict__ Y,
                                int total) {
    int i = blockIdx.x * blockDim.x + threadIdx.x;
    if (i < total) {
        int n = i >> 8;            // /256
        int j = i & (DFEAT - 1);
        float di = dinv[n];
        Y[i] = bias[j] + di * di * G[i];
    }
}

// ---------- edge scatter: Y[col] += dinv[row]*dinv[col] * G[row], one wave per edge ----------
__global__ __launch_bounds__(256)
void scatter_edges_kernel(const float* __restrict__ G, const int* __restrict__ row,
                          const int* __restrict__ col, const float* __restrict__ dinv,
                          float* __restrict__ Y, int e) {
    int gid  = blockIdx.x * blockDim.x + threadIdx.x;
    int wave = gid >> 5;
    int lane = gid & 31;
    if (wave >= e) return;
    int r = row[wave];
    int c = col[wave];
    float coeff = dinv[r] * dinv[c];
    const float4* src = (const float4*)(G + (size_t)r * DFEAT);
    float* dst = Y + (size_t)c * DFEAT;
#pragma unroll
    for (int it = 0; it < 2; ++it) {
        int idx = lane + it * 32;          // 64 float4 = 256 floats
        float4 v = src[idx];               // global_load_b128
        int base = idx * 4;
        atomicAdd(dst + base + 0, coeff * v.x);
        atomicAdd(dst + base + 1, coeff * v.y);
        atomicAdd(dst + base + 2, coeff * v.z);
        atomicAdd(dst + base + 3, coeff * v.w);
    }
}

static inline size_t align256(size_t x) { return (x + 255) & ~(size_t)255; }

extern "C" void kernel_launch(void* const* d_in, const int* in_sizes, int n_in,
                              void* d_out, int out_size, void* d_ws, size_t ws_size,
                              hipStream_t stream) {
    const float* x    = (const float*)d_in[0];
    const int*   eidx = (const int*)d_in[1];
    const float* Wl[3] = { (const float*)d_in[2], (const float*)d_in[4], (const float*)d_in[6] };
    const float* bl[3] = { (const float*)d_in[3], (const float*)d_in[5], (const float*)d_in[7] };

    const int N = in_sizes[0] / DFEAT;     // 50000
    const int E = in_sizes[1] / 2;         // 800000
    const int* rowp = eidx;                // edge_index[0] = source (gather)
    const int* colp = eidx + E;            // edge_index[1] = target (scatter)

    // workspace carve-out
    char* ws = (char*)d_ws;
    size_t off = 0;
    float*  dinv = (float*)(ws + off);  off = align256(off + (size_t)N * 4);
    __bf16* Hbf  = (__bf16*)(ws + off); off = align256(off + (size_t)N * DFEAT * 2);
    __bf16* Wbf  = (__bf16*)(ws + off); off = align256(off + (size_t)DFEAT * DFEAT * 2);
    float*  G    = (float*)(ws + off);  off = align256(off + (size_t)N * DFEAT * 4);
    float*  Y    = (float*)d_out;       // aggregation target every layer; final layer = output

    const int total = N * DFEAT;
    const int TB = 256;

    // normalization: deg -> dinv
    deg_init_kernel<<<(N + TB - 1) / TB, TB, 0, stream>>>(dinv, N);
    deg_accum_kernel<<<(E + TB - 1) / TB, TB, 0, stream>>>(colp, dinv, E);
    deg_rsqrt_kernel<<<(N + TB - 1) / TB, TB, 0, stream>>>(dinv, N);

    const int mTiles     = (N + 15) / 16;                    // 3125
    const int mBlocks    = (mTiles + 7) / 8;                 // 391
    const int gemmBlocks = mBlocks * 4;                      // x 4 N-groups
    const int edgeBlocks = (E * 32 + TB - 1) / TB;           // 100000 blocks

    for (int l = 0; l < 3; ++l) {
        // input features -> bf16 (ReLU on previous layer's output for l>0)
        const float* src = (l == 0) ? x : Y;
        cvt_bf16_kernel<<<(total + TB - 1) / TB, TB, 0, stream>>>(src, Hbf, total, l > 0 ? 1 : 0);
        // weights -> bf16
        cvt_bf16_kernel<<<(DFEAT * DFEAT + TB - 1) / TB, TB, 0, stream>>>(Wl[l], Wbf,
                                                                          DFEAT * DFEAT, 0);
        // G = Hbf @ Wbf^T  (f32 accum via v_wmma_f32_16x16x32_bf16; LDS-staged B,
        // double-buffered fragments)
        gemm_wmma_bf16_kernel<<<gemmBlocks, TB, 0, stream>>>(Hbf, Wbf, G, N);
        // Y = bias + dinv^2 * G   (self-loop, initializes Y)
        init_out_kernel<<<(total + TB - 1) / TB, TB, 0, stream>>>(G, dinv, bl[l], Y, total);
        // Y[col] += dinv[row]*dinv[col]*G[row] over all edges
        scatter_edges_kernel<<<edgeBlocks, TB, 0, stream>>>(G, rowp, colp, dinv, Y, E);
    }
}
